// GTN_14542759264256
// MI455X (gfx1250) — matrix operand order repaired and checked by reading, not compiled, not run on hardware
//
#include <hip/hip_runtime.h>
#include <hip/hip_bf16.h>
#include <math.h>

#define D_NUM  1500
#define P_NUM  4500
#define N_ALL  6000
#define EMB    128
#define LAYERS 3

typedef __attribute__((ext_vector_type(16))) __bf16 v16bf;
typedef __attribute__((ext_vector_type(8)))  __bf16 v8bf;
typedef __attribute__((ext_vector_type(8)))  float  v8f;

static inline int ceil_div(int a, int b) { return (a + b - 1) / b; }
static inline int r32(int x) { return (x + 31) & ~31; }
static inline int r64(int x) { return (x + 63) & ~63; }

// =====================================================================
// WMMA GEMM:  C[M,N] (store | +=atomic)  A[M,K] * op(B)[K,N]
// Both operands are pre-prepared, zero-padded bf16 panels:
//   A  : [r64(M)][r32(K)]  row-major            (lda  = r32(K))
//   BT : [Npad ][r32(K)]  op(B) column-major,   (ldbt = r32(K))
//        with cs[k] (sym-norm dinv_k) folded in during prep
// -> hot loop is 2 + 2*NT unguarded global_load_b128 + NT wmma, nothing
//    else. rs[m] (dinv_i / 1/cnt) applied in the guarded epilogue.
// mode: 0=store, 1=store+bias, 2=store+bias+relu, 3=atomicAdd (split-K)
// =====================================================================
template<int NT>
__global__ __launch_bounds__(128)
void gemm_kernel(float* __restrict__ C,
                 const __bf16* __restrict__ A,
                 const __bf16* __restrict__ BT,
                 int M, int N, int K, int lda, int ldbt, int ldc,
                 const float* __restrict__ rs,
                 const float* __restrict__ bias,
                 int mode, int kChunk)
{
    const int tid  = threadIdx.x;
    const int wave = tid >> 5;
    const int lane = tid & 31;
    const int half = lane >> 4;
    const int lm   = lane & 15;

    const int rowBase = (blockIdx.x * 4 + wave) * 16;
    if (rowBase >= M) return;                     // wave-uniform, no barriers
    const int colBase = blockIdx.z * (NT * 16);
    const int Kround  = (K + 31) & ~31;
    const int kBegin  = blockIdx.y * kChunk;
    const int kEnd    = min(Kround, kBegin + kChunk);

    const __bf16* ap = A + (size_t)(rowBase + lm) * lda;

    v8f acc[NT];
    #pragma unroll
    for (int t = 0; t < NT; ++t)
        #pragma unroll
        for (int i = 0; i < 8; ++i) acc[t][i] = 0.0f;

    for (int k0 = kBegin; k0 < kEnd; k0 += 32) {
        // A fragment (ISA 7.12.2 16-bit A layout): two b128 loads
        v8bf a0 = *(const v8bf*)(ap + k0 + half * 8);
        v8bf a1 = *(const v8bf*)(ap + k0 + 16 + half * 8);
        v16bf af;
        #pragma unroll
        for (int j = 0; j < 8; ++j) { af[j] = a0[j]; af[8 + j] = a1[j]; }

        // B fragments: two b128 loads per tile, then WMMA
        #pragma unroll
        for (int t = 0; t < NT; ++t) {
            const __bf16* bp = BT + (size_t)(colBase + t * 16 + lm) * ldbt
                                  + k0 + half * 8;
            v8bf b0 = *(const v8bf*)bp;
            v8bf b1 = *(const v8bf*)(bp + 16);
            v16bf bf;
            #pragma unroll
            for (int j = 0; j < 8; ++j) { bf[j] = b0[j]; bf[8 + j] = b1[j]; }
            acc[t] = __builtin_amdgcn_wmma_f32_16x16x32_bf16(
                false, af, false, bf, (short)0, acc[t], false, false);
        }
    }

    // epilogue (only guards in the kernel)
    #pragma unroll
    for (int t = 0; t < NT; ++t) {
        const int col = colBase + t * 16 + lm;
        if (col >= N) continue;
        #pragma unroll
        for (int r = 0; r < 8; ++r) {
            const int row = rowBase + half * 8 + r;
            if (row >= M) continue;
            float v = acc[t][r];
            if (rs != nullptr) v *= rs[row];
            float* p = &C[(size_t)row * ldc + col];
            if (mode == 3) {
                atomicAdd(p, v);
            } else {
                if (mode >= 1 && bias != nullptr) v += bias[col];
                if (mode == 2) v = v > 0.0f ? v : 0.0f;
                *p = v;
            }
        }
    }
}

// =====================================================================
// Panel preparation kernels
// =====================================================================
// f32 -> zero-padded bf16 A panel (optionally transposed source)
__global__ void cvt_bf16_kernel(__bf16* __restrict__ dst,
                                const float* __restrict__ src,
                                int rows, int cols, int ldsrc,
                                int colsPad, int total, int trans)
{
    int idx = blockIdx.x * blockDim.x + threadIdx.x;
    if (idx >= total) return;
    const int r = idx / colsPad;
    const int c = idx - r * colsPad;
    float v = 0.0f;
    if (r < rows && c < cols)
        v = trans ? src[(size_t)c * ldsrc + r] : src[(size_t)r * ldsrc + c];
    dst[idx] = (__bf16)v;
}

// BT[n][k] = op(B)[k][n] * cs[k]  (btrans: op(B)[k][n] = B[n*ldb+k])
__global__ void prep_bt_kernel(__bf16* __restrict__ bt,
                               const float* __restrict__ B,
                               const float* __restrict__ cs,
                               int Ncols, int K, int ldb, int Kpad,
                               int total, int btrans)
{
    int idx = blockIdx.x * blockDim.x + threadIdx.x;
    if (idx >= total) return;
    const int n = idx / Kpad;
    const int k = idx - n * Kpad;
    float v = 0.0f;
    if (n < Ncols && k < K) {
        float b = btrans ? B[(size_t)n * ldb + k] : B[(size_t)k * ldb + n];
        v = b * cs[k];
    }
    bt[idx] = (__bf16)v;
}

// =====================================================================
// Small helper kernels
// =====================================================================
__global__ void rowsum_kernel(const float* __restrict__ A, int lda,
                              int rows, int cols, int trans, int mode,
                              float* __restrict__ out)
{
    __shared__ float red[256];
    const int i = blockIdx.x;
    if (i >= rows) return;
    float s = 0.0f;
    for (int j = threadIdx.x; j < cols; j += blockDim.x)
        s += trans ? A[(size_t)j * lda + i] : A[(size_t)i * lda + j];
    red[threadIdx.x] = s;
    __syncthreads();
    for (int off = 128; off > 0; off >>= 1) {
        if (threadIdx.x < off) red[threadIdx.x] += red[threadIdx.x + off];
        __syncthreads();
    }
    if (threadIdx.x == 0) {
        float d = red[0];
        out[i] = (d > 0.0f) ? (mode == 0 ? rsqrtf(d) : 1.0f / d) : 0.0f;
    }
}

__global__ void zero_kernel(float* p, int n) {
    int i = blockIdx.x * blockDim.x + threadIdx.x;
    if (i < n) p[i] = 0.0f;
}

__global__ void fill_ones_kernel(float* p, int len, int K) {
    int i = blockIdx.x * blockDim.x + threadIdx.x;
    if (i < len) p[i] = (i < K) ? 1.0f : 0.0f;
}

__global__ void axpby_kernel(float* __restrict__ out,
                             const float* __restrict__ x,
                             const float* __restrict__ y,
                             float a, float b, int n) {
    int i = blockIdx.x * blockDim.x + threadIdx.x;
    if (i < n) out[i] = a * x[i] + (y ? b * y[i] : 0.0f);
}

__global__ void add_inplace_kernel(float* __restrict__ out,
                                   const float* __restrict__ x, int n) {
    int i = blockIdx.x * blockDim.x + threadIdx.x;
    if (i < n) out[i] += x[i];
}

__global__ void dot_h_kernel(const float* __restrict__ X,
                             const float* __restrict__ h,
                             float* __restrict__ out, int n) {
    __shared__ float red[128];
    const int i = blockIdx.x;
    if (i >= n) return;
    const int c = threadIdx.x;
    red[c] = X[(size_t)i * EMB + c] * h[c];
    __syncthreads();
    for (int off = 64; off > 0; off >>= 1) {
        if (c < off) red[c] += red[c + off];
        __syncthreads();
    }
    if (c == 0) out[i] = red[0];
}

__global__ void log_softmax_kernel(float* __restrict__ t, int n) {
    __shared__ float red[256];
    float m = -3.402823e38f;
    for (int i = threadIdx.x; i < n; i += blockDim.x) m = fmaxf(m, t[i]);
    red[threadIdx.x] = m;
    __syncthreads();
    for (int off = 128; off > 0; off >>= 1) {
        if (threadIdx.x < off)
            red[threadIdx.x] = fmaxf(red[threadIdx.x], red[threadIdx.x + off]);
        __syncthreads();
    }
    m = red[0];
    __syncthreads();
    float s = 0.0f;
    for (int i = threadIdx.x; i < n; i += blockDim.x) s += expf(t[i] - m);
    red[threadIdx.x] = s;
    __syncthreads();
    for (int off = 128; off > 0; off >>= 1) {
        if (threadIdx.x < off) red[threadIdx.x] += red[threadIdx.x + off];
        __syncthreads();
    }
    const float lse = m + logf(red[0]);
    __syncthreads();
    for (int i = threadIdx.x; i < n; i += blockDim.x) t[i] -= lse;
}

__global__ void combine_drug_kernel(float* __restrict__ fin,
                                    const float* __restrict__ di,
                                    const float* __restrict__ dr,
                                    const float* __restrict__ ds,
                                    const float* __restrict__ w1,
                                    const float* __restrict__ w2,
                                    const float* __restrict__ w3, int n) {
    int idx = blockIdx.x * blockDim.x + threadIdx.x;
    if (idx >= n * EMB) return;
    int i = idx >> 7;
    float dw = w1[i], drw = w2[i], dsw = w3[i];
    float a = dw / (dw + drw + dsw);
    float b = drw / (a + drw + dsw);
    float c = 1.0f - a - b;
    fin[idx] = a * di[idx] + b * dr[idx] + c * ds[idx];
}

__global__ void combine_pro_kernel(float* __restrict__ fin,
                                   const float* __restrict__ pi_,
                                   const float* __restrict__ pr,
                                   const float* __restrict__ w1,
                                   const float* __restrict__ w2, int n) {
    int idx = blockIdx.x * blockDim.x + threadIdx.x;
    if (idx >= n * EMB) return;
    int i = idx >> 7;
    float pw = w1[i], prw = w2[i];
    float pa = pw / (pw + prw);
    fin[idx] = pa * pi_[idx] + (1.0f - pa) * pr[idx];
}

__global__ void zero_d_kernel(double* p, int n) {
    int i = blockIdx.x * blockDim.x + threadIdx.x;
    if (i < n) p[i] = 0.0;
}

__global__ void stats_kernel(const float* __restrict__ y, int n,
                             double* __restrict__ acc) {
    __shared__ double s1[256], s2[256];
    double a = 0.0, b = 0.0;
    for (int i = blockIdx.x * blockDim.x + threadIdx.x; i < n;
         i += gridDim.x * blockDim.x) {
        double v = (double)y[i];
        a += v; b += v * v;
    }
    s1[threadIdx.x] = a; s2[threadIdx.x] = b;
    __syncthreads();
    for (int off = 128; off > 0; off >>= 1) {
        if (threadIdx.x < off) {
            s1[threadIdx.x] += s1[threadIdx.x + off];
            s2[threadIdx.x] += s2[threadIdx.x + off];
        }
        __syncthreads();
    }
    if (threadIdx.x == 0) {
        atomicAdd(&acc[0], s1[0]);
        atomicAdd(&acc[1], s2[0]);
    }
}

__global__ void finalize_kernel(float* __restrict__ y, int n,
                                const double* __restrict__ acc) {
    int i = blockIdx.x * blockDim.x + threadIdx.x;
    if (i >= n) return;
    double mean = acc[0] / (double)n;
    double var  = (acc[1] - (double)n * mean * mean) / (double)(n - 1);
    double inv  = 1.0 / sqrt(var);
    float z = (float)(((double)y[i] - mean) * inv);
    y[i] = 1.0f / (1.0f + expf(-z));
}

// =====================================================================
// Host-side helpers
// =====================================================================
static void zero_f(hipStream_t st, float* p, int n) {
    zero_kernel<<<ceil_div(n, 256), 256, 0, st>>>(p, n);
}
static void copy_f(hipStream_t st, float* dst, const float* src, int n) {
    axpby_kernel<<<ceil_div(n, 256), 256, 0, st>>>(dst, src, nullptr, 1.0f, 0.0f, n);
}

// prep BT then launch GEMM (A already a bf16 panel with lda = r32(K))
static void gemm_bt(hipStream_t st, float* C, const __bf16* Ap,
                    const float* Bsrc, int ldb, int btrans, const float* cs,
                    int M, int N, int K, int ldc,
                    const float* rs, const float* bias,
                    int mode, int chunks, __bf16* bt)
{
    const int Kp = r32(K);
    const int nt = (N > 128) ? 4 : 8;
    const int CT = nt * 16;
    const int Npad = ceil_div(N, CT) * CT;
    const int tot = Npad * Kp;
    prep_bt_kernel<<<ceil_div(tot, 256), 256, 0, st>>>(bt, Bsrc, cs, N, K, ldb, Kp, tot, btrans);

    int kChunk = Kp;
    if (chunks > 1) kChunk = r32(ceil_div(Kp, chunks));
    dim3 grid(ceil_div(ceil_div(M, 16), 4), ceil_div(Kp, kChunk), ceil_div(N, CT));
    if (nt == 8)
        gemm_kernel<8><<<grid, 128, 0, st>>>(C, Ap, bt, M, N, K, Kp, Kp, ldc, rs, bias, mode, kChunk);
    else
        gemm_kernel<4><<<grid, 128, 0, st>>>(C, Ap, bt, M, N, K, Kp, Kp, ldc, rs, bias, mode, kChunk);
}

// convert an f32 A operand into a bf16 panel
static const __bf16* prep_a(hipStream_t st, const float* Asrc, int M, int K,
                            int lda, __bf16* panel)
{
    const int Mp = r64(M), Kp = r32(K);
    const int tot = Mp * Kp;
    cvt_bf16_kernel<<<ceil_div(tot, 256), 256, 0, st>>>(panel, Asrc, M, K, lda, Kp, tot, 0);
    return panel;
}

// GCN: out = (x0 + e1 + e2 + e3)/4 over sym-normed adjacency (bf16 panel)
static void run_gcn(hipStream_t st, const __bf16* Ab, const float* dinv, int sz,
                    const float* x0, float* dst, float* eA, float* eB,
                    float* acc, int chunks, __bf16* bt)
{
    const int n = sz * EMB;
    copy_f(st, acc, x0, n);
    const float* cur = x0;
    float* nxt = eA;
    for (int l = 0; l < LAYERS; ++l) {
        zero_f(st, nxt, n);
        gemm_bt(st, nxt, Ab, cur, EMB, 0, dinv, sz, EMB, sz, EMB,
                dinv, nullptr, 3, chunks, bt);
        add_inplace_kernel<<<ceil_div(n, 256), 256, 0, st>>>(acc, nxt, n);
        cur = nxt;
        nxt = (nxt == eA) ? eB : eA;
    }
    axpby_kernel<<<ceil_div(n, 256), 256, 0, st>>>(dst, acc, nullptr, 0.25f, 0.0f, n);
}

// Bipartite GCN over sym_norm([0 R; R^T 0]); acc <- x0 + e1 + e2 + e3
static void run_gcn_bip(hipStream_t st, const __bf16* Rb, const __bf16* RTb,
                        const float* dinvRr, const float* dinvRc,
                        const float* x0, float* acc, float* eA, float* eB,
                        __bf16* bt)
{
    const int n = N_ALL * EMB;
    copy_f(st, acc, x0, n);
    const float* cur = x0;
    float* nxt = eA;
    for (int l = 0; l < LAYERS; ++l) {
        zero_f(st, nxt, n);
        gemm_bt(st, nxt, Rb, cur + (size_t)D_NUM * EMB, EMB, 0, dinvRc,
                D_NUM, EMB, P_NUM, EMB, dinvRr, nullptr, 3, 8, bt);
        gemm_bt(st, nxt + (size_t)D_NUM * EMB, RTb, cur, EMB, 0, dinvRr,
                P_NUM, EMB, D_NUM, EMB, dinvRc, nullptr, 3, 4, bt);
        add_inplace_kernel<<<ceil_div(n, 256), 256, 0, st>>>(acc, nxt, n);
        cur = nxt;
        nxt = (nxt == eA) ? eB : eA;
    }
}

// =====================================================================
// Entry point
// =====================================================================
extern "C" void kernel_launch(void* const* d_in, const int* in_sizes, int n_in,
                              void* d_out, int out_size, void* d_ws, size_t ws_size,
                              hipStream_t stream)
{
    (void)in_sizes; (void)n_in; (void)out_size; (void)ws_size;
    const float* Af          = (const float*)d_in[0];
    const float* drug_struct = (const float*)d_in[1];
    const float* prot_struct = (const float*)d_in[2];
    const float* lin_drug_w  = (const float*)d_in[3];
    const float* lin_drug_b  = (const float*)d_in[4];
    const float* lin_pro_w   = (const float*)d_in[5];
    const float* lin_pro_b   = (const float*)d_in[6];
    const float* p_weight    = (const float*)d_in[7];
    const float* d_weight_i  = (const float*)d_in[8];
    const float* pd_weight_p = (const float*)d_in[9];
    const float* pd_weight_d = (const float*)d_in[10];
    const float* dp_weight_p = (const float*)d_in[11];
    const float* WA_drug = (const float*)d_in[12];
    const float* BA_drug = (const float*)d_in[13];
    const float* HA_drug = (const float*)d_in[14];
    const float* WB_drug = (const float*)d_in[15];
    const float* BB_drug = (const float*)d_in[16];
    const float* HB_drug = (const float*)d_in[17];
    const float* WA_pro  = (const float*)d_in[18];
    const float* BA_pro  = (const float*)d_in[19];
    const float* HA_pro  = (const float*)d_in[20];
    const float* WB_pro  = (const float*)d_in[21];
    const float* BB_pro  = (const float*)d_in[22];
    const float* HB_pro  = (const float*)d_in[23];
    const float* WA_sim  = (const float*)d_in[24];
    const float* BA_sim  = (const float*)d_in[25];
    const float* HA_sim  = (const float*)d_in[26];
    float* OUT = (float*)d_out;

    const size_t NN = (size_t)N_ALL * N_ALL;
    const float* A2 = Af + 2 * NN;                                   // drug-drug
    const float* A3 = Af + 3 * NN + (size_t)D_NUM * N_ALL + D_NUM;   // prot-prot
    const float* A4 = Af + 4 * NN;                                   // drug-sim
    const float* Rp = Af + D_NUM;                                    // R = A[0,:D,D:]

    const int Dp64 = r64(D_NUM), Pp64 = r64(P_NUM);        // 1536 / 4544
    const int Dp32 = r32(D_NUM), Pp32 = r32(P_NUM);        // 1504 / 4512
    const int DEp = Dp64 * EMB, PEp = Pp64 * EMB;
    const int NEp = (N_ALL + 80) * EMB;

    // ---- workspace carve-up (byte allocator, 256B aligned) ----
    char* base = (char*)d_ws;
    size_t off = 0;
    auto allocF = [&](size_t n) { float*  p = (float*) (base + off); off += (n * 4 + 255) & ~(size_t)255; return p; };
    auto allocH = [&](size_t n) { __bf16* p = (__bf16*)(base + off); off += (n * 2 + 255) & ~(size_t)255; return p; };

    __bf16* A2b = allocH((size_t)Dp64 * Dp32);
    __bf16* A3b = allocH((size_t)Pp64 * Pp32);
    __bf16* A4b = allocH((size_t)Dp64 * Dp32);
    __bf16* Rb  = allocH((size_t)Dp64 * Pp32);
    __bf16* RTb = allocH((size_t)Pp64 * Dp32);
    __bf16* atS = allocH((size_t)Pp64 * 512);   // A-panel scratch (max 4544x512)
    __bf16* btS = allocH((size_t)Pp64 * 160);   // BT scratch (max 4544x128 / 128x4512)

    float* dru_str     = allocF(DEp);
    float* pro_str     = allocF(PEp);
    float* x0d         = allocF(DEp);
    float* x0p         = allocF(PEp);
    float* eA          = allocF(NEp);
    float* eB          = allocF(NEp);
    float* accG        = allocF(NEp);
    float* Xbuf        = allocF(NEp);
    float* dru_rel_emb = allocF(DEp);
    float* dru_rel_nei = allocF(DEp);
    float* pro_rel_emb = allocF(PEp);
    float* pro_rel_nei = allocF(PEp);
    float* dru_sim_emb = allocF(DEp);
    float* dru_sim_nei = allocF(DEp);
    float* dru_tem     = allocF(DEp);
    float* pro_tem     = allocF(PEp);
    float* one_all     = allocF(DEp);
    float* two_allA    = allocF(PEp);
    float* two_allB    = allocF(PEp);
    float* one_emb     = allocF(DEp);
    float* two_emb     = allocF(PEp);
    float* dru_int     = allocF(DEp);
    float* pro_int     = allocF(PEp);
    float* gw_tmp      = allocF(PEp);
    float* fin_dru     = allocF(DEp);
    float* fin_pro     = allocF(PEp);
    float* dinv2  = allocF(Dp64);  float* ic2  = allocF(Dp64);
    float* dinv3  = allocF(Pp64);  float* ic3  = allocF(Pp64);
    float* dinv4  = allocF(Dp64);  float* ic4  = allocF(Dp64);
    float* icRr   = allocF(Dp64);  float* icRc = allocF(Pp64);
    float* dinvRr = allocF(Dp64);  float* dinvRc = allocF(Pp64);
    float* onesD  = allocF(Dp64);  float* onesP = allocF(Pp64);
    float* ones512 = allocF(512);
    float* wdrug  = allocF(Dp64);  float* wpro  = allocF(Pp64);
    float* wdrel  = allocF(Dp64);  float* wprel = allocF(Pp64);
    float* wdsim  = allocF(Dp64);
    double* stats = (double*)allocF(8);

    // zero whole used workspace: all padding becomes true 0.0
    zero_f(stream, (float*)base, (int)(off / 4));

    // ---- ones / scale vectors ----
    fill_ones_kernel<<<ceil_div(Dp64, 256), 256, 0, stream>>>(onesD, Dp64, D_NUM);
    fill_ones_kernel<<<ceil_div(Pp64, 256), 256, 0, stream>>>(onesP, Pp64, P_NUM);
    fill_ones_kernel<<<2, 256, 0, stream>>>(ones512, 512, 512);

    rowsum_kernel<<<D_NUM, 256, 0, stream>>>(A2, N_ALL, D_NUM, D_NUM, 0, 0, dinv2);
    rowsum_kernel<<<D_NUM, 256, 0, stream>>>(A2, N_ALL, D_NUM, D_NUM, 0, 1, ic2);
    rowsum_kernel<<<P_NUM, 256, 0, stream>>>(A3, N_ALL, P_NUM, P_NUM, 0, 0, dinv3);
    rowsum_kernel<<<P_NUM, 256, 0, stream>>>(A3, N_ALL, P_NUM, P_NUM, 0, 1, ic3);
    rowsum_kernel<<<D_NUM, 256, 0, stream>>>(A4, N_ALL, D_NUM, D_NUM, 0, 0, dinv4);
    rowsum_kernel<<<D_NUM, 256, 0, stream>>>(A4, N_ALL, D_NUM, D_NUM, 0, 1, ic4);
    rowsum_kernel<<<D_NUM, 256, 0, stream>>>(Rp, N_ALL, D_NUM, P_NUM, 0, 1, icRr);
    rowsum_kernel<<<D_NUM, 256, 0, stream>>>(Rp, N_ALL, D_NUM, P_NUM, 0, 0, dinvRr);
    rowsum_kernel<<<P_NUM, 256, 0, stream>>>(Rp, N_ALL, P_NUM, D_NUM, 1, 1, icRc);
    rowsum_kernel<<<P_NUM, 256, 0, stream>>>(Rp, N_ALL, P_NUM, D_NUM, 1, 0, dinvRc);

    // ---- adjacency -> zero-padded bf16 panels (halves HBM bytes) ----
    {
        int tot;
        tot = Dp64 * Dp32;
        cvt_bf16_kernel<<<ceil_div(tot, 256), 256, 0, stream>>>(A2b, A2, D_NUM, D_NUM, N_ALL, Dp32, tot, 0);
        cvt_bf16_kernel<<<ceil_div(tot, 256), 256, 0, stream>>>(A4b, A4, D_NUM, D_NUM, N_ALL, Dp32, tot, 0);
        tot = Pp64 * Pp32;
        cvt_bf16_kernel<<<ceil_div(tot, 256), 256, 0, stream>>>(A3b, A3, P_NUM, P_NUM, N_ALL, Pp32, tot, 0);
        tot = Dp64 * Pp32;
        cvt_bf16_kernel<<<ceil_div(tot, 256), 256, 0, stream>>>(Rb, Rp, D_NUM, P_NUM, N_ALL, Pp32, tot, 0);
        tot = Pp64 * Dp32;
        cvt_bf16_kernel<<<ceil_div(tot, 256), 256, 0, stream>>>(RTb, Rp, P_NUM, D_NUM, N_ALL, Dp32, tot, 1);
    }

    // ---- input linear layers ----
    prep_a(stream, drug_struct, D_NUM, 160, 160, atS);
    gemm_bt(stream, dru_str, atS, lin_drug_w, EMB, 0, ones512,
            D_NUM, EMB, 160, EMB, nullptr, lin_drug_b, 1, 1, btS);
    prep_a(stream, prot_struct, P_NUM, 512, 512, atS);
    gemm_bt(stream, pro_str, atS, lin_pro_w, EMB, 0, ones512,
            P_NUM, EMB, 512, EMB, nullptr, lin_pro_b, 1, 1, btS);

    // ---- x0 = feats @ w ----
    prep_a(stream, dru_str, D_NUM, EMB, EMB, atS);
    gemm_bt(stream, x0d, atS, d_weight_i, EMB, 0, ones512,
            D_NUM, EMB, EMB, EMB, nullptr, nullptr, 0, 1, btS);
    prep_a(stream, pro_str, P_NUM, EMB, EMB, atS);
    gemm_bt(stream, x0p, atS, p_weight, EMB, 0, ones512,
            P_NUM, EMB, EMB, EMB, nullptr, nullptr, 0, 1, btS);

    // ---- relational GCNs + neighbor means ----
    run_gcn(stream, A2b, dinv2, D_NUM, x0d, dru_rel_emb, eA, eB, accG, 4, btS);
    zero_f(stream, dru_rel_nei, D_NUM * EMB);
    gemm_bt(stream, dru_rel_nei, A2b, dru_str, EMB, 0, onesD,
            D_NUM, EMB, D_NUM, EMB, ic2, nullptr, 3, 4, btS);

    run_gcn(stream, A3b, dinv3, P_NUM, x0p, pro_rel_emb, eA, eB, accG, 8, btS);
    zero_f(stream, pro_rel_nei, P_NUM * EMB);
    gemm_bt(stream, pro_rel_nei, A3b, pro_str, EMB, 0, onesP,
            P_NUM, EMB, P_NUM, EMB, ic3, nullptr, 3, 8, btS);

    run_gcn(stream, A4b, dinv4, D_NUM, x0d, dru_sim_emb, eA, eB, accG, 4, btS);
    zero_f(stream, dru_sim_nei, D_NUM * EMB);
    gemm_bt(stream, dru_sim_nei, A4b, dru_str, EMB, 0, onesD,
            D_NUM, EMB, D_NUM, EMB, ic4, nullptr, 3, 4, btS);

    // ---- cross-domain neighbor means ----
    zero_f(stream, dru_tem, D_NUM * EMB);
    gemm_bt(stream, dru_tem, Rb, pro_rel_nei, EMB, 0, onesP,
            D_NUM, EMB, P_NUM, EMB, icRr, nullptr, 3, 8, btS);
    zero_f(stream, pro_tem, P_NUM * EMB);
    gemm_bt(stream, pro_tem, RTb, dru_rel_nei, EMB, 0, onesD,
            P_NUM, EMB, D_NUM, EMB, icRc, nullptr, 3, 4, btS);

    // ---- fused features ----
    const int DE = D_NUM * EMB, PE = P_NUM * EMB;
    axpby_kernel<<<ceil_div(DE, 256), 256, 0, stream>>>(one_all, dru_str, dru_tem, 0.8f, 0.2f, DE);
    axpby_kernel<<<ceil_div(PE, 256), 256, 0, stream>>>(two_allA, pro_str, pro_tem, 0.8f, 0.2f, PE);
    prep_a(stream, two_allA, P_NUM, EMB, EMB, atS);
    gemm_bt(stream, two_allB, atS, dp_weight_p, EMB, 0, ones512,
            P_NUM, EMB, EMB, EMB, nullptr, nullptr, 0, 1, btS);
    prep_a(stream, two_allB, P_NUM, EMB, EMB, atS);
    gemm_bt(stream, two_allA, atS, pd_weight_d, EMB, 0, ones512,
            P_NUM, EMB, EMB, EMB, nullptr, nullptr, 0, 1, btS);
    prep_a(stream, dru_str, D_NUM, EMB, EMB, atS);
    gemm_bt(stream, one_emb, atS, pd_weight_d, EMB, 0, ones512,
            D_NUM, EMB, EMB, EMB, nullptr, nullptr, 0, 1, btS);
    prep_a(stream, pro_str, P_NUM, EMB, EMB, atS);
    gemm_bt(stream, two_emb, atS, pd_weight_p, EMB, 0, ones512,
            P_NUM, EMB, EMB, EMB, nullptr, nullptr, 0, 1, btS);

    // ---- bipartite GCN #1 -> dru_int ----
    copy_f(stream, Xbuf, one_emb, DE);
    copy_f(stream, Xbuf + (size_t)D_NUM * EMB, two_allA, PE);
    run_gcn_bip(stream, Rb, RTb, dinvRr, dinvRc, Xbuf, accG, eA, eB, btS);
    axpby_kernel<<<ceil_div(DE, 256), 256, 0, stream>>>(dru_int, accG, nullptr, 0.25f, 0.0f, DE);

    // ---- bipartite GCN #2 -> pro_int ----
    copy_f(stream, Xbuf, one_all, DE);
    copy_f(stream, Xbuf + (size_t)D_NUM * EMB, two_emb, PE);
    run_gcn_bip(stream, Rb, RTb, dinvRr, dinvRc, Xbuf, accG, eA, eB, btS);
    axpby_kernel<<<ceil_div(PE, 256), 256, 0, stream>>>(pro_int, accG + (size_t)D_NUM * EMB,
                                                        nullptr, 0.25f, 0.0f, PE);

    // ---- attention weights: log_softmax(relu(emb@W + B) @ H) ----
    auto get_w = [&](const float* emb, int n, const float* WW, const float* BB,
                     const float* HH, float* wout) {
        prep_a(stream, emb, n, EMB, EMB, atS);
        gemm_bt(stream, gw_tmp, atS, WW, EMB, 0, ones512,
                n, EMB, EMB, EMB, nullptr, BB, 2, 1, btS);
        dot_h_kernel<<<n, 128, 0, stream>>>(gw_tmp, HH, wout, n);
        log_softmax_kernel<<<1, 256, 0, stream>>>(wout, n);
    };
    get_w(dru_int,     D_NUM, WA_drug, BA_drug, HA_drug, wdrug);
    get_w(pro_int,     P_NUM, WA_pro,  BA_pro,  HA_pro,  wpro);
    get_w(dru_rel_emb, D_NUM, WB_drug, BB_drug, HB_drug, wdrel);
    get_w(pro_rel_emb, P_NUM, WB_pro,  BB_pro,  HB_pro,  wprel);
    get_w(dru_sim_emb, D_NUM, WA_sim,  BA_sim,  HA_sim,  wdsim);

    // ---- weighted fusion ----
    combine_drug_kernel<<<ceil_div(DE, 256), 256, 0, stream>>>(
        fin_dru, dru_int, dru_rel_emb, dru_sim_emb, wdrug, wdrel, wdsim, D_NUM);
    combine_pro_kernel<<<ceil_div(PE, 256), 256, 0, stream>>>(
        fin_pro, pro_int, pro_rel_emb, wpro, wprel, P_NUM);

    // ---- score GEMM: y = fin_dru @ fin_pro^T (BT = fin_pro row-major) ----
    prep_a(stream, fin_dru, D_NUM, EMB, EMB, atS);
    gemm_bt(stream, OUT, atS, fin_pro, EMB, 1, ones512,
            D_NUM, P_NUM, EMB, P_NUM, nullptr, nullptr, 0, 1, btS);

    // ---- standardize (ddof=1) + sigmoid ----
    const int ny = D_NUM * P_NUM;
    zero_d_kernel<<<1, 32, 0, stream>>>(stats, 2);
    stats_kernel<<<256, 256, 0, stream>>>(OUT, ny, stats);
    finalize_kernel<<<ceil_div(ny, 256), 256, 0, stream>>>(OUT, ny, stats);
}
